// TransformerEncoderLayer_27255862460847
// MI455X (gfx1250) — matrix operand order repaired and checked by reading, not compiled, not run on hardware
//
#include <hip/hip_runtime.h>
#include <hip/hip_bf16.h>
#include <math.h>

// ---------------- types ----------------
typedef __bf16 bf16_t;
typedef __attribute__((ext_vector_type(8)))  __bf16 v8bf;
typedef __attribute__((ext_vector_type(16))) __bf16 v16bf;
typedef __attribute__((ext_vector_type(8)))  float  v8f;
typedef __attribute__((ext_vector_type(4)))  int    i4v;
typedef __attribute__((address_space(1))) i4v* as1_i4p;   // global int4*
typedef __attribute__((address_space(3))) i4v* as3_i4p;   // LDS int4*

// Problem constants (from reference)
#define BB 2
#define SS 2048
#define DD 768
#define HH 12
#define HD 64
#define FF 3072
#define MM (BB * SS)       // 4096 rows
#define NQKV (3 * DD)      // 2304

// ---------------- optional CDNA5 async global->LDS path ----------------
#if defined(__has_builtin)
#if __has_builtin(__builtin_amdgcn_global_load_async_to_lds_b128)
#define USE_ASYNC_LDS 1
#endif
#if __has_builtin(__builtin_amdgcn_s_wait_asynccnt)
#define HAVE_WAIT_ASYNC_BUILTIN 1
#endif
#endif
#ifndef USE_ASYNC_LDS
#define USE_ASYNC_LDS 0
#endif
#ifndef HAVE_WAIT_ASYNC_BUILTIN
#define HAVE_WAIT_ASYNC_BUILTIN 0
#endif

// Copy 16 bytes global -> LDS. Async (ASYNCcnt-tracked) when available.
__device__ __forceinline__ void cp16_g2l(bf16_t* ldst, const bf16_t* gsrc) {
#if USE_ASYNC_LDS
  __builtin_amdgcn_global_load_async_to_lds_b128(
      (as1_i4p)(uintptr_t)gsrc,
      (as3_i4p)(unsigned)(uintptr_t)ldst,   // low 32 bits of generic = LDS offset
      0, 0);
#else
  *reinterpret_cast<v8bf*>(ldst) = *reinterpret_cast<const v8bf*>(gsrc);
#endif
}

__device__ __forceinline__ void wait_async_lds() {
#if USE_ASYNC_LDS
#if HAVE_WAIT_ASYNC_BUILTIN
  __builtin_amdgcn_s_wait_asynccnt(0);
#else
  asm volatile("s_wait_asynccnt 0" ::: "memory");
#endif
#endif
}

// ---------------- helpers ----------------
__device__ __forceinline__ v8f zero8() {
  v8f z;
  #pragma unroll
  for (int i = 0; i < 8; ++i) z[i] = 0.0f;
  return z;
}

// Assemble a 16-element bf16 fragment from two 16-byte chunks.
__device__ __forceinline__ v16bf ld16(const bf16_t* p0, const bf16_t* p1) {
  v8bf lo = *reinterpret_cast<const v8bf*>(p0);
  v8bf hi = *reinterpret_cast<const v8bf*>(p1);
  return __builtin_shufflevector(lo, hi, 0,1,2,3,4,5,6,7,8,9,10,11,12,13,14,15);
}

__device__ __forceinline__ v8f wmma_bf16(v16bf a, v16bf b, v8f c) {
  // (neg_a, A, neg_b, B, c_mod, C, reuse_a, reuse_b)
  return __builtin_amdgcn_wmma_f32_16x16x32_bf16(false, a, false, b, (short)0, c, false, false);
}

__device__ __forceinline__ float gelu_exact(float x) {
  return 0.5f * x * (1.0f + erff(x * 0.70710678118654752440f));
}

// ---------------- weight/bias repack kernels ----------------

// wq/wk/wv: [H, D, HD] f32 -> fused bf16 [D, 3*D] (cols: q | k | v, each h*64+e)
__global__ void pack_qkv_w(const float* __restrict__ wq, const float* __restrict__ wk,
                           const float* __restrict__ wv, bf16_t* __restrict__ W) {
  int idx = blockIdx.x * blockDim.x + threadIdx.x;   // idx = h*D*HD + d*HD + e
  if (idx >= HH * DD * HD) return;
  int h = idx / (DD * HD);
  int r = idx % (DD * HD);
  int d = r / HD;
  int e = r % HD;
  int col = h * HD + e;
  W[(size_t)d * NQKV + col]            = (bf16_t)wq[idx];
  W[(size_t)d * NQKV + DD + col]       = (bf16_t)wk[idx];
  W[(size_t)d * NQKV + 2 * DD + col]   = (bf16_t)wv[idx];
}

// bq/bk/bv: [H, HD] f32 -> fused f32 [2304]
__global__ void pack_qkv_b(const float* __restrict__ bq, const float* __restrict__ bk,
                           const float* __restrict__ bv, float* __restrict__ bqkv) {
  int idx = blockIdx.x * blockDim.x + threadIdx.x;
  if (idx >= NQKV) return;
  int sec = idx / DD;
  int w = idx % DD;          // = h*64 + e, matching [H,HD] flat
  const float* src = (sec == 0) ? bq : (sec == 1) ? bk : bv;
  bqkv[idx] = src[w];
}

__global__ void cvt_f32_bf16(const float* __restrict__ src, bf16_t* __restrict__ dst, int n) {
  int i = blockIdx.x * blockDim.x + threadIdx.x;
  if (i < n) dst[i] = (bf16_t)src[i];
}

// ---------------- LayerNorm -> bf16 ----------------
__global__ __launch_bounds__(256) void ln_to_bf16(const float* __restrict__ x,
                                                  const float* __restrict__ g,
                                                  const float* __restrict__ be,
                                                  bf16_t* __restrict__ out) {
  int row = blockIdx.x;
  int t = threadIdx.x;
  const float* xr = x + (size_t)row * DD;
  float v0 = xr[t], v1 = xr[t + 256], v2 = xr[t + 512];
  __shared__ float r1[256], r2[256];
  r1[t] = v0 + v1 + v2;
  r2[t] = v0 * v0 + v1 * v1 + v2 * v2;
  __syncthreads();
  #pragma unroll
  for (int off = 128; off > 0; off >>= 1) {
    if (t < off) { r1[t] += r1[t + off]; r2[t] += r2[t + off]; }
    __syncthreads();
  }
  float mu = r1[0] * (1.0f / DD);
  float var = r2[0] * (1.0f / DD) - mu * mu;
  float rstd = rsqrtf(var + 1e-5f);
  bf16_t* orow = out + (size_t)row * DD;
  orow[t]       = (bf16_t)((v0 - mu) * rstd * g[t]       + be[t]);
  orow[t + 256] = (bf16_t)((v1 - mu) * rstd * g[t + 256] + be[t + 256]);
  orow[t + 512] = (bf16_t)((v2 - mu) * rstd * g[t + 512] + be[t + 512]);
}

// ---------------- WMMA GEMM (double-buffered LDS, async global->LDS) ----------------
// C[M,N] = epilogue( A[M,K](bf16) @ W[K,N](bf16) + bias ), optional residual / gelu.
// Block: 256 threads (8 wave32), tile 128x128, k-step 32.
template <bool GELU, bool RESID, bool WF32, bool WBF16>
__global__ __launch_bounds__(256) void gemm_bf16_wmma(
    const bf16_t* __restrict__ A, const bf16_t* __restrict__ W,
    const float* __restrict__ bias, const float* __restrict__ R,
    float* __restrict__ outF, bf16_t* __restrict__ outB,
    int M, int N, int K) {
  constexpr int BM = 128, BN = 128, BK = 32;
  constexpr int LDA = 40;    // bf16 units, 80B rows (16B aligned, conflict-spread)
  constexpr int LDB = 136;   // bf16 units, 272B rows
  __shared__ bf16_t sA[2][BM * LDA];
  __shared__ bf16_t sB[2][BK * LDB];

  const int t = threadIdx.x;
  const int m0 = blockIdx.y * BM, n0 = blockIdx.x * BN;
  const int lane = t & 31, wave = t >> 5;
  const int wm = wave & 1, wn = wave >> 1;       // 2 x 4 wave grid
  const int lr = lane & 15, lh = lane >> 4;

  v8f acc[4][2];
  #pragma unroll
  for (int i = 0; i < 4; ++i)
    #pragma unroll
    for (int j = 0; j < 2; ++j) acc[i][j] = zero8();

  const int arow = t >> 1, acol = (t & 1) * 16;  // A: 128 rows x 32 cols, 16 bf16/thread
  const int brow = t >> 3, bcol = (t & 7) * 16;  // B: 32 rows x 128 cols

  auto stage_tile = [&](int k0, int buf) {
    const bf16_t* ga = A + (size_t)(m0 + arow) * K + k0 + acol;
    cp16_g2l(&sA[buf][arow * LDA + acol], ga);
    cp16_g2l(&sA[buf][arow * LDA + acol + 8], ga + 8);
    const bf16_t* gb = W + (size_t)(k0 + brow) * N + n0 + bcol;
    cp16_g2l(&sB[buf][brow * LDB + bcol], gb);
    cp16_g2l(&sB[buf][brow * LDB + bcol + 8], gb + 8);
    if (k0 + BK < K) {
      __builtin_prefetch(ga + BK, 0, 0);                     // global_prefetch_b8
      __builtin_prefetch(gb + (size_t)BK * N, 0, 0);
    }
  };

  stage_tile(0, 0);

  const int nk = K / BK;
  for (int ki = 0; ki < nk; ++ki) {
    const int buf = ki & 1;
    wait_async_lds();
    __syncthreads();                       // tile ki visible; prev compute done
    if (ki + 1 < nk) stage_tile((ki + 1) * BK, buf ^ 1);

    v16bf af[4], bfr[2];
    #pragma unroll
    for (int i = 0; i < 4; ++i) {
      // A 16x32 bf16 fragment: lane row = lr, K halves 8*lh and 16+8*lh
      const bf16_t* p = &sA[buf][(wm * 64 + i * 16 + lr) * LDA + 8 * lh];
      af[i] = ld16(p, p + 16);
    }
    #pragma unroll
    for (int j = 0; j < 2; ++j) {
      // B 32x16 bf16 fragment: lane col = lr, K halves 8*lh and 16+8*lh
      const bf16_t* p = &sB[buf][wn * 32 + j * 16 + lr];
      #pragma unroll
      for (int s = 0; s < 16; ++s) bfr[j][s] = p[((s & 7) + 8 * lh + 16 * (s >> 3)) * LDB];
    }
    #pragma unroll
    for (int i = 0; i < 4; ++i)
      #pragma unroll
      for (int j = 0; j < 2; ++j) acc[i][j] = wmma_bf16(af[i], bfr[j], acc[i][j]);
  }

  // Epilogue (C layout: VGPR v -> row v + 8*lh; col = lane%16 within tile)
  #pragma unroll
  for (int i = 0; i < 4; ++i) {
    #pragma unroll
    for (int j = 0; j < 2; ++j) {
      int col = n0 + wn * 32 + j * 16 + lr;
      float bv = bias[col];
      #pragma unroll
      for (int v = 0; v < 8; ++v) {
        int row = m0 + wm * 64 + i * 16 + v + 8 * lh;
        float val = acc[i][j][v] + bv;
        if constexpr (RESID) val += R[(size_t)row * N + col];
        if constexpr (GELU) val = gelu_exact(val);
        if constexpr (WF32)  outF[(size_t)row * N + col] = val;
        if constexpr (WBF16) outB[(size_t)row * N + col] = (bf16_t)val;
      }
    }
  }
}

// ---------------- Flash attention (bf16 WMMA, fp32 online softmax) ----------------
// QKV: [B*S, 2304] bf16 (q | k | v). Out: attn [B*S, 768] bf16.
// Block: 128 threads = 4 waves; each wave owns 16 query rows; block covers 64.
__global__ __launch_bounds__(128) void attn_flash(const bf16_t* __restrict__ QKV,
                                                  bf16_t* __restrict__ attnO) {
  const int t = threadIdx.x, lane = t & 31, wave = t >> 5;
  const int lr = lane & 15, lh = lane >> 4;
  const int qblk = blockIdx.x, h = blockIdx.y, b = blockIdx.z;

  __shared__ bf16_t Kt[64 * 72];      // [hd][key], transposed K tile
  __shared__ bf16_t Vt[64 * 72];      // [key][hd]
  __shared__ float  Sc[4][16 * 68];   // per-wave score tile (16 rows x 64 keys)
  __shared__ bf16_t Pb[4][16 * 72];   // per-wave probabilities (bf16)
  __shared__ float  rowM[64], rowL[64], rowA[64];

  if (t < 64) { rowM[t] = -1e30f; rowL[t] = 0.0f; rowA[t] = 0.0f; }
  __syncthreads();

  const int qbase = qblk * 64 + wave * 16;

  // Q fragments (constant over key blocks): A-layout, row = lr
  v16bf qf[2];
  const bf16_t* qrow = QKV + (size_t)(b * SS + qbase + lr) * NQKV + h * HD;
  #pragma unroll
  for (int kc = 0; kc < 2; ++kc) {
    const bf16_t* p = qrow + kc * 32 + 8 * lh;
    qf[kc] = ld16(p, p + 16);
  }

  v8f o[4];
  #pragma unroll
  for (int i = 0; i < 4; ++i) o[i] = zero8();

  for (int kb = 0; kb < SS; kb += 64) {
    // ---- stage K (transposed, via VGPR) and V (async row copy) tiles ----
    {
      int key = t >> 1, hb = (t & 1) * 32;
      const bf16_t* vp = QKV + (size_t)(b * SS + kb + key) * NQKV + 2 * DD + h * HD + hb;
      #pragma unroll
      for (int c = 0; c < 4; ++c)
        cp16_g2l(&Vt[key * 72 + hb + c * 8], vp + c * 8);   // async when available
      const bf16_t* kp = QKV + (size_t)(b * SS + kb + key) * NQKV + DD + h * HD + hb;
      bf16_t tmp[32];
      #pragma unroll
      for (int c = 0; c < 4; ++c)
        *reinterpret_cast<v8bf*>(&tmp[c * 8]) = *reinterpret_cast<const v8bf*>(kp + c * 8);
      #pragma unroll
      for (int jj = 0; jj < 32; ++jj) Kt[(hb + jj) * 72 + key] = tmp[jj];
    }
    wait_async_lds();
    __syncthreads();

    // ---- scores: Q(16x64) @ K^T(64x64) ----
    v8f sacc[4];
    #pragma unroll
    for (int i = 0; i < 4; ++i) sacc[i] = zero8();
    #pragma unroll
    for (int tile = 0; tile < 4; ++tile) {
      #pragma unroll
      for (int kc = 0; kc < 2; ++kc) {
        const bf16_t* p = &Kt[(kc * 32) * 72 + tile * 16 + lr];
        v16bf kf;
        #pragma unroll
        for (int s = 0; s < 16; ++s) kf[s] = p[((s & 7) + 8 * lh + 16 * (s >> 3)) * 72];
        sacc[tile] = wmma_bf16(qf[kc], kf, sacc[tile]);
      }
    }
    #pragma unroll
    for (int tile = 0; tile < 4; ++tile)
      #pragma unroll
      for (int v = 0; v < 8; ++v)
        Sc[wave][(v + 8 * lh) * 68 + tile * 16 + lr] = sacc[tile][v];

    // ---- online softmax, one row per lane (lanes 0..15) ----
    if (lane < 16) {
      float* sr = &Sc[wave][lane * 68];
      int ri = wave * 16 + lane;
      float mo = rowM[ri];
      float mx = mo;
      #pragma unroll 8
      for (int c = 0; c < 64; ++c) {
        float sv = sr[c] * 0.125f;   // 1/sqrt(64)
        sr[c] = sv;
        mx = fmaxf(mx, sv);
      }
      float al = __expf(mo - mx);
      float sum = 0.0f;
      bf16_t* pr = &Pb[wave][lane * 72];
      #pragma unroll 8
      for (int c = 0; c < 64; ++c) {
        float e = __expf(sr[c] - mx);
        sum += e;
        pr[c] = (bf16_t)e;
      }
      rowM[ri] = mx;
      rowL[ri] = rowL[ri] * al + sum;
      rowA[ri] = al;
    }

    // ---- rescale running output, then O += P(16x64) @ V(64x64) ----
    float alv[8];
    #pragma unroll
    for (int v = 0; v < 8; ++v) alv[v] = rowA[wave * 16 + v + 8 * lh];
    #pragma unroll
    for (int tile = 0; tile < 4; ++tile)
      #pragma unroll
      for (int v = 0; v < 8; ++v) o[tile][v] *= alv[v];

    #pragma unroll
    for (int tile = 0; tile < 4; ++tile) {
      #pragma unroll
      for (int kc = 0; kc < 2; ++kc) {
        const bf16_t* pa = &Pb[wave][lr * 72 + kc * 32 + 8 * lh];
        const bf16_t* pb = &Vt[(kc * 32) * 72 + tile * 16 + lr];
        v16bf vf;
        #pragma unroll
        for (int s = 0; s < 16; ++s) vf[s] = pb[((s & 7) + 8 * lh + 16 * (s >> 3)) * 72];
        o[tile] = wmma_bf16(ld16(pa, pa + 16), vf, o[tile]);
      }
    }
    __syncthreads();
  }

  // ---- normalize and write bf16 attention output ----
  float li[8];
  #pragma unroll
  for (int v = 0; v < 8; ++v) li[v] = 1.0f / rowL[wave * 16 + v + 8 * lh];
  #pragma unroll
  for (int tile = 0; tile < 4; ++tile) {
    #pragma unroll
    for (int v = 0; v < 8; ++v) {
      int q = qbase + v + 8 * lh;
      int col = h * HD + tile * 16 + lr;
      attnO[(size_t)(b * SS + q) * DD + col] = (bf16_t)(o[tile][v] * li[v]);
    }
  }
}

// ---------------- host-side orchestration ----------------
extern "C" void kernel_launch(void* const* d_in, const int* in_sizes, int n_in,
                              void* d_out, int out_size, void* d_ws, size_t ws_size,
                              hipStream_t stream) {
  (void)in_sizes; (void)n_in; (void)out_size; (void)ws_size;
  const float* x   = (const float*)d_in[0];
  const float* wq  = (const float*)d_in[1];
  const float* bq  = (const float*)d_in[2];
  const float* wk  = (const float*)d_in[3];
  const float* bk  = (const float*)d_in[4];
  const float* wv  = (const float*)d_in[5];
  const float* bv  = (const float*)d_in[6];
  const float* wo  = (const float*)d_in[7];
  const float* bo  = (const float*)d_in[8];
  const float* g1  = (const float*)d_in[9];
  const float* be1 = (const float*)d_in[10];
  const float* g2  = (const float*)d_in[11];
  const float* be2 = (const float*)d_in[12];
  const float* w1  = (const float*)d_in[13];
  const float* bf1 = (const float*)d_in[14];
  const float* w2  = (const float*)d_in[15];
  const float* bf2 = (const float*)d_in[16];

  // carve workspace (256B aligned slices)
  char* base = (char*)d_ws;
  size_t off = 0;
  auto alloc = [&](size_t bytes) -> char* {
    char* p = base + off;
    off = (off + bytes + 255) & ~(size_t)255;
    return p;
  };
  bf16_t* h1    = (bf16_t*)alloc((size_t)MM * DD * 2);
  bf16_t* Wqkv  = (bf16_t*)alloc((size_t)DD * NQKV * 2);
  float*  bqkv  = (float*) alloc((size_t)NQKV * 4);
  bf16_t* QKV   = (bf16_t*)alloc((size_t)MM * NQKV * 2);
  bf16_t* attnb = (bf16_t*)alloc((size_t)MM * DD * 2);
  bf16_t* wob   = (bf16_t*)alloc((size_t)DD * DD * 2);
  float*  x1    = (float*) alloc((size_t)MM * DD * 4);
  bf16_t* h2    = (bf16_t*)alloc((size_t)MM * DD * 2);
  bf16_t* w1b   = (bf16_t*)alloc((size_t)DD * FF * 2);
  bf16_t* a1    = (bf16_t*)alloc((size_t)MM * FF * 2);
  bf16_t* w2b   = (bf16_t*)alloc((size_t)FF * DD * 2);

  // 1) weight/bias repack to bf16
  pack_qkv_w<<<(HH * DD * HD + 255) / 256, 256, 0, stream>>>(wq, wk, wv, Wqkv);
  pack_qkv_b<<<(NQKV + 255) / 256, 256, 0, stream>>>(bq, bk, bv, bqkv);
  cvt_f32_bf16<<<(DD * DD + 255) / 256, 256, 0, stream>>>(wo, wob, DD * DD);
  cvt_f32_bf16<<<(DD * FF + 255) / 256, 256, 0, stream>>>(w1, w1b, DD * FF);
  cvt_f32_bf16<<<(FF * DD + 255) / 256, 256, 0, stream>>>(w2, w2b, FF * DD);

  // 2) LN1
  ln_to_bf16<<<MM, 256, 0, stream>>>(x, g1, be1, h1);

  // 3) fused QKV projection: [4096x768] @ [768x2304] -> bf16
  gemm_bf16_wmma<false, false, false, true>
      <<<dim3(NQKV / 128, MM / 128), 256, 0, stream>>>(h1, Wqkv, bqkv, nullptr,
                                                       nullptr, QKV, MM, NQKV, DD);

  // 4) attention
  attn_flash<<<dim3(SS / 64, HH, BB), 128, 0, stream>>>(QKV, attnb);

  // 5) output projection + residual(x) -> x1 (f32)
  gemm_bf16_wmma<false, true, true, false>
      <<<dim3(DD / 128, MM / 128), 256, 0, stream>>>(attnb, wob, bo, x,
                                                     x1, nullptr, MM, DD, DD);

  // 6) LN2
  ln_to_bf16<<<MM, 256, 0, stream>>>(x1, g2, be2, h2);

  // 7) FFN up + exact GELU -> bf16
  gemm_bf16_wmma<true, false, false, true>
      <<<dim3(FF / 128, MM / 128), 256, 0, stream>>>(h2, w1b, bf1, nullptr,
                                                     nullptr, a1, MM, FF, DD);

  // 8) FFN down + residual(x1) -> d_out (f32)
  gemm_bf16_wmma<false, true, true, false>
      <<<dim3(DD / 128, MM / 128), 256, 0, stream>>>(a1, w2b, bf2, x1,
                                                     (float*)d_out, nullptr, MM, DD, FF);
}
